// HEGNN_20633022890062
// MI455X (gfx1250) — matrix operand order repaired and checked by reading, not compiled, not run on hardware
//
#include <hip/hip_runtime.h>
#include <hip/hip_bf16.h>
#include <stdint.h>

// ---------------------------------------------------------------------------
// Problem constants (fixed by the reference)
// ---------------------------------------------------------------------------
constexpr int NN = 20000;     // nodes
constexpr int EE = 320000;    // edges
constexpr int GG = 8;         // graphs
constexpr int DD = 128;       // feature dim

// ---------------------------------------------------------------------------
// WMMA types / helpers
// ---------------------------------------------------------------------------
typedef __attribute__((ext_vector_type(16))) __bf16 v16bf;
typedef __attribute__((ext_vector_type(8)))  float  v8f;

union FragU {
  uint16_t u[16];
  uint4    q[2];
  v16bf    v;
};

__device__ inline v8f v8f_zero() {
  v8f z = {0.f, 0.f, 0.f, 0.f, 0.f, 0.f, 0.f, 0.f};
  return z;
}

__device__ inline v8f wmma_bf16(v16bf a, v16bf b, v8f c) {
  return __builtin_amdgcn_wmma_f32_16x16x32_bf16(
      /*neg_a=*/false, a, /*neg_b=*/false, b,
      /*c_mod=*/(short)0, c, /*reuse_a=*/false, /*reuse_b=*/false);
}

__device__ inline uint16_t f2bf(float f) {
  uint32_t u = __builtin_bit_cast(uint32_t, f);
  uint32_t r = u + 0x7FFFu + ((u >> 16) & 1u);
  return (uint16_t)(r >> 16);
}

__device__ inline float siluf(float x) { return x / (1.0f + __expf(-x)); }

__device__ inline float artanh_clip(float v) {
  v = fminf(fmaxf(v, -1.0f + 1e-5f), 1.0f - 1e-5f);
  return 0.5f * __logf((1.0f + v) / (1.0f - v));
}

// DS ops from one wave execute in-order in the LDS pipe; this fence drains the
// counter and is a compiler memory barrier so staged tiles are visible.
__device__ inline void lds_fence() {
  asm volatile("s_wait_dscnt 0" ::: "memory");
}

// ---------------------------------------------------------------------------
// Async global -> LDS copy (CDNA5 GLOBAL_LOAD_ASYNC_TO_LDS_B128, ASYNCcnt).
// Per-lane: LDS[lds_addr] = MEM[gaddr] for 16 bytes. Generic LDS pointers on
// amdgcn carry the LDS byte offset in the low 32 bits.
// ---------------------------------------------------------------------------
__device__ inline uint32_t lds_off32(const void* p) {
  return (uint32_t)(uintptr_t)p;
}
__device__ inline void async_load_b128(uint32_t lds_addr, const void* gaddr) {
  asm volatile("global_load_async_to_lds_b128 %0, %1, off"
               :
               : "v"(lds_addr), "v"(gaddr)
               : "memory");
}
__device__ inline void wait_async0() {
  asm volatile("s_wait_asynccnt 0" ::: "memory");
}

// A-fragment (16x32 bf16) gathered from a row-major LDS tile.
// Layout per ISA 7.12.2: lane -> M = lane&15 ; K-half select by lane bit4.
// Element runs are contiguous 8x u16, so this lowers to 2x ds_load_b128.
__device__ inline v16bf loadA_lds(const uint16_t* buf, int stride, int lane, int kbase) {
  FragU f;
  const int m = lane & 15;
  const int khalf = (lane & 16) ? 8 : 0;
  const uint16_t* row = buf + m * stride + kbase + khalf;
#pragma unroll
  for (int e = 0; e < 16; ++e) {
    int v4 = e >> 1;
    int k = ((v4 & 4) << 2) + ((v4 & 3) << 1) + (e & 1);  // (v>=4 ? 16:0) + (v%4)*2 + sub
    f.u[e] = row[k];
  }
  return f.v;
}

// B-fragment: each lane reads 32 contiguous bytes (global or LDS pointer).
__device__ inline v16bf loadB(const uint16_t* tile, int lane) {
  FragU f;
  const uint4* p = reinterpret_cast<const uint4*>(tile) + (lane << 1);
  f.q[0] = p[0];
  f.q[1] = p[1];
  return f.v;
}

// ---------------------------------------------------------------------------
// Weight pre-swizzle: dst[(nt*KT+kt)*512 + lane*16 + e] = src[n, k] (bf16),
// with n = nt*16 + (lane&15), k = kt*32 + (e>>3)*16 + (lane>>4)*8 + (e&7) ; zero-padded.
// ---------------------------------------------------------------------------
__global__ void pack_b_kernel(const float* __restrict__ src, uint16_t* __restrict__ dst,
                              int nRows, int srcStride, int kUsed, int NT, int KT) {
  int total = NT * KT * 512;
  for (int idx = blockIdx.x * blockDim.x + threadIdx.x; idx < total;
       idx += gridDim.x * blockDim.x) {
    int e = idx & 15;
    int lane = (idx >> 4) & 31;
    int tile = idx >> 9;
    int kt = tile % KT, nt = tile / KT;
    int n = nt * 16 + (lane & 15);
    int k = kt * 32 + (e >> 3) * 16 + (lane >> 4) * 8 + (e & 7);
    float val = (n < nRows && k < kUsed) ? src[(long)n * srcStride + k] : 0.0f;
    dst[idx] = f2bf(val);
  }
}

// Precombine output matmuls: Wcomb = W_out @ W_f2 (4x128); bcomb = W_out@b_f2 + b_out
__global__ void comb_kernel(const float* __restrict__ W_f2, const float* __restrict__ W_out,
                            const float* __restrict__ b_f2, const float* __restrict__ b_out,
                            float* __restrict__ Wcomb, float* __restrict__ bcomb) {
  int tid = blockIdx.x * blockDim.x + threadIdx.x;
  if (tid < 512) {
    int o = tid >> 7, d = tid & 127;
    float s = 0.f;
    for (int k = 0; k < 128; ++k) s += W_out[o * 128 + k] * W_f2[k * 128 + d];
    Wcomb[o * 128 + d] = s;
  }
  if (tid < 4) {
    float s = b_out[tid];
    for (int k = 0; k < 128; ++k) s += W_out[tid * 128 + k] * b_f2[k];
    bcomb[tid] = s;
  }
}

// ---------------------------------------------------------------------------
// Time embedding
// ---------------------------------------------------------------------------
__global__ void time_kernel(const float* __restrict__ tin,
                            const float* __restrict__ W_t1, const float* __restrict__ b_t1,
                            const float* __restrict__ W_t2, const float* __restrict__ b_t2,
                            const float* __restrict__ W_tl, const float* __restrict__ b_tl,
                            float* __restrict__ te) {
  __shared__ float t1s[128];
  __shared__ float t2s[128];
  int tid = threadIdx.x;
  for (int g = 0; g < GG; ++g) {
    if (tid < 128) t1s[tid] = siluf(tin[g] * W_t1[tid] + b_t1[tid]);
    __syncthreads();
    if (tid < 128) {
      float s = b_t2[tid];
      for (int k = 0; k < 128; ++k) s += t1s[k] * W_t2[tid * 128 + k];
      t2s[tid] = s;
    }
    __syncthreads();
    {
      float s = b_tl[tid];
      for (int k = 0; k < 128; ++k) s += siluf(t2s[k]) * W_tl[tid * 128 + k];
      te[g * 256 + tid] = s;
    }
    __syncthreads();
  }
}

// ---------------------------------------------------------------------------
// Hyperbolic node features (one 128-thread block per node)
// ---------------------------------------------------------------------------
__global__ __launch_bounds__(128) void node_feats_kernel(
    const float* __restrict__ x, const float* __restrict__ ex,
    const float* __restrict__ W_hyp, float* __restrict__ feats,
    uint16_t* __restrict__ featsbf) {
  __shared__ float xs[16];
  __shared__ float red[128];
  const int node = blockIdx.x;
  const int tid = threadIdx.x;

  if (tid < 11) xs[tid] = x[(long)node * 11 + tid];
  else if (tid < 15) xs[tid] = ex[(long)node * 4 + (tid - 11)];
  __syncthreads();

  float n2 = 0.f;
#pragma unroll
  for (int k = 0; k < 15; ++k) n2 += xs[k] * xs[k];
  float nr = sqrtf(n2);
  float n = fmaxf(nr, 1e-15f);
  float psc = tanhf(n) / n;                 // expmap0 scale
  float pnorm = psc * nr;                   // |p|
  float projs = (pnorm > 0.996f) ? (0.996f / fmaxf(pnorm, 1e-15f)) : 1.0f;
  psc *= projs;
  float xnc = fmaxf(pnorm * projs, 1e-15f); // clipped |x_h|

  float mx = 0.f;
#pragma unroll
  for (int k = 0; k < 15; ++k) mx += W_hyp[tid * 15 + k] * xs[k];
  mx *= psc;

  red[tid] = mx * mx;
  __syncthreads();
  for (int s = 64; s > 0; s >>= 1) {
    if (tid < s) red[tid] += red[tid + s];
    __syncthreads();
  }
  float mxn2 = red[0];
  float mxn_r = sqrtf(mxn2);
  float mxn = fmaxf(mxn_r, 1e-15f);

  float at = artanh_clip(xnc);
  float th = tanhf(mxn / xnc * at);
  float resc = (mxn2 == 0.f) ? 0.f : th / mxn;     // res = resc * mx
  float rn = fabsf(th) * (mxn_r / mxn);            // |res|
  float projs2 = (rn > 0.996f) ? (0.996f / fmaxf(rn, 1e-15f)) : 1.0f;
  float fn = fmaxf(rn * projs2, 1e-15f);
  float lsc = artanh_clip(fn) / fn;                // logmap0 scale

  float val = mx * resc * projs2 * lsc;
  feats[(long)node * DD + tid] = val;
  featsbf[(long)node * DD + tid] = f2bf(val);
}

// ---------------------------------------------------------------------------
// Fused per-edge pipeline with block-shared, async-DMA'd, double-buffered
// weight staging in LDS. One wave per 16 edges; block = 4 waves (64 edges).
// ---------------------------------------------------------------------------
constexpr int EDGE_WAVE_BYTES  = 17920;  // per-wave: 12288 minB + 1024 h1t + 4096 mijB + rel/cw/node
constexpr int EDGE_STAGE_OFF   = 4 * EDGE_WAVE_BYTES;  // 71680
constexpr int EDGE_STAGE_BYTES = 32768;  // 32 tiles x 1KB per buffer (x2 buffers)

// Stage one chunk's B tiles: 24 contiguous GEMM1 tiles (waves 0-2) and the 8
// strided GEMM2 tiles (wave 3). 256B per thread, via async global->LDS DMA.
__device__ inline void stage_chunk(const uint16_t* We1p, const uint16_t* We2p,
                                   char* sBuf, int c, int tid) {
  if (tid < 96) {
    const char* src = (const char*)(We1p + ((long)(2 * c * 12) << 9)) + tid * 256;
    uint32_t dst = lds_off32(sBuf) + tid * 256;
#pragma unroll
    for (int q = 0; q < 16; ++q) async_load_b128(dst + q * 16, src + q * 16);
  } else {
    int t = (tid - 96) >> 2;
    int sub = (tid - 96) & 3;
    const char* src = (const char*)(We2p + ((long)(t * 25 + c) << 9)) + sub * 256;
    uint32_t dst = lds_off32(sBuf) + (24 + t) * 1024 + sub * 256;
#pragma unroll
    for (int q = 0; q < 16; ++q) async_load_b128(dst + q * 16, src + q * 16);
  }
}

// Stage one group (8 nt2 x 4 kt = 32 contiguous tiles) of the W_c1 GEMM.
__device__ inline void stage_cw(const uint16_t* Wc1p, char* sBuf, int g, int tid) {
  const char* src = (const char*)Wc1p + (long)g * 32768 + tid * 256;
  uint32_t dst = lds_off32(sBuf) + tid * 256;
#pragma unroll
  for (int q = 0; q < 16; ++q) async_load_b128(dst + q * 16, src + q * 16);
}

__global__ __launch_bounds__(128) void edge_kernel(
    const float* __restrict__ pos, const float* __restrict__ ear,
    const int* __restrict__ eidx,
    const float* __restrict__ W_ee, const float* __restrict__ b_ee,
    const float* __restrict__ W_e1, const float* __restrict__ b_e1,
    const float* __restrict__ b_e2, const float* __restrict__ b_c1,
    const float* __restrict__ W_c2, const float* __restrict__ b_c2,
    const uint16_t* __restrict__ featsbf,
    const uint16_t* __restrict__ We1p, const uint16_t* __restrict__ We2p,
    const uint16_t* __restrict__ Wc1p,
    float* __restrict__ m_i, float* __restrict__ num, float* __restrict__ cnt) {
  extern __shared__ char smem[];
  const int tid = threadIdx.x;
  const int lane = tid & 31;
  const int wave = tid >> 5;
  char* wbase = smem + wave * EDGE_WAVE_BYTES;
  uint16_t* minB = (uint16_t*)(wbase);              // 16 x 384 bf16 (row major)
  uint16_t* h1t  = (uint16_t*)(wbase + 12288);      // 16 x 32 bf16
  uint16_t* mijB = (uint16_t*)(wbase + 13312);      // 16 x 128 bf16
  float*    relB = (float*)(wbase + 17408);         // 16 x {rx,ry,rz,rdist}
  float*    cwB  = (float*)(wbase + 17664);         // 16
  int*      nodeB= (int*)(wbase + 17728);           // 16 x {i,j}
  char*     stage = smem + EDGE_STAGE_OFF;          // 2 x 32KB shared B staging

  const long eb = ((long)blockIdx.x * 4 + wave) * 16;
  const int m16 = lane & 15;
  const int hf = lane >> 4;
  const int nlo = lane & 15;
  const int Mrow = hf * 8;  // C-layout: M = v + Mrow, N = nlo

  // ---- per-edge geometry + node ids -------------------------------------
  if (lane < 16) {
    long e = eb + lane;
    int i = eidx[e];
    int j = eidx[(long)EE + e];
    nodeB[lane * 2] = i;
    nodeB[lane * 2 + 1] = j;
    float rx = pos[(long)i * 3 + 0] - pos[(long)j * 3 + 0];
    float ry = pos[(long)i * 3 + 1] - pos[(long)j * 3 + 1];
    float rz = pos[(long)i * 3 + 2] - pos[(long)j * 3 + 2];
    relB[lane * 4 + 0] = rx;
    relB[lane * 4 + 1] = ry;
    relB[lane * 4 + 2] = rz;
    relB[lane * 4 + 3] = rx * rx + ry * ry + rz * rz;
    cwB[lane] = 0.0f;
  }
  lds_fence();

  // ---- stage m_in tile (async DMA) + chunk-0 weights --------------------
  {
    int i = nodeB[m16 * 2], j = nodeB[m16 * 2 + 1];
    const char* si = (const char*)(featsbf + (long)i * DD + hf * 64);
    const char* sj = (const char*)(featsbf + (long)j * DD + hf * 64);
    uint32_t di = lds_off32(minB + m16 * 384 + hf * 64);
    uint32_t dj = lds_off32(minB + m16 * 384 + 128 + hf * 64);
#pragma unroll
    for (int q = 0; q < 8; ++q) {
      async_load_b128(di + q * 16, si + q * 16);
      async_load_b128(dj + q * 16, sj + q * 16);
    }
  }
  stage_chunk(We1p, We2p, stage, 0, tid);
  {
    long e = eb + m16;
    float a0 = ear[e * 4 + 0], a1 = ear[e * 4 + 1], a2 = ear[e * 4 + 2], a3 = ear[e * 4 + 3];
    for (int nOut = hf * 64; nOut < hf * 64 + 64; ++nOut) {
      float v = b_ee[nOut] + a0 * W_ee[nOut * 4 + 0] + a1 * W_ee[nOut * 4 + 1] +
                a2 * W_ee[nOut * 4 + 2] + a3 * W_ee[nOut * 4 + 3];
      minB[m16 * 384 + 256 + nOut] = f2bf(v);
    }
  }
  wait_async0();
  lds_fence();
  __syncthreads();

  // ---- preload all GEMM1 A fragments into registers (invariant) ----------
  v16bf aF[12];
#pragma unroll
  for (int kt = 0; kt < 12; ++kt) aF[kt] = loadA_lds(minB, 384, lane, kt * 32);

  // ---- fused GEMM1 (K=384) / GEMM2 (K=800) over 25 hidden chunks ---------
  v8f mij[8];
#pragma unroll
  for (int t = 0; t < 8; ++t) mij[t] = v8f_zero();

  for (int c = 0; c < 25; ++c) {
    // kick off DMA of the next chunk's weights while computing this one
    if (c < 24) stage_chunk(We1p, We2p, stage + ((c + 1) & 1) * EDGE_STAGE_BYTES, c + 1, tid);
    const uint16_t* sb = (const uint16_t*)(stage + (c & 1) * EDGE_STAGE_BYTES);

    v8f acc0 = v8f_zero(), acc1 = v8f_zero();
#pragma unroll
    for (int kt = 0; kt < 12; ++kt) {
      acc0 = wmma_bf16(aF[kt], loadB(sb + (long)kt * 512, lane), acc0);
      acc1 = wmma_bf16(aF[kt], loadB(sb + (long)(12 + kt) * 512, lane), acc1);
    }
    // bias + rank-1 rdist column + silu -> transpose slab through LDS
#pragma unroll
    for (int t = 0; t < 2; ++t) {
      int ntile = 2 * c + t;
      int n = ntile * 16 + nlo;
      float bn = (n < 770) ? b_e1[n] : 0.0f;
      float wrd = (n < 770) ? W_e1[(long)n * 385 + 384] : 0.0f;
      const v8f acc = t ? acc1 : acc0;
#pragma unroll
      for (int v = 0; v < 8; ++v) {
        int m = v + Mrow;
        float val = acc[v] + bn + relB[m * 4 + 3] * wrd;
        h1t[m * 32 + t * 16 + nlo] = f2bf(siluf(val));
      }
    }
    lds_fence();
    v16bf a2 = loadA_lds(h1t, 32, lane, 0);
#pragma unroll
    for (int t = 0; t < 8; ++t)
      mij[t] = wmma_bf16(a2, loadB(sb + (long)(24 + t) * 512, lane), mij[t]);

    wait_async0();     // next chunk's DMA landed
    __syncthreads();   // all waves done with current buffer before reuse
  }

  // ---- m_ij epilogue: silu + scatter m_i + stage for cw-MLP --------------
#pragma unroll
  for (int t = 0; t < 8; ++t) {
    int n = t * 16 + nlo;
    float bn = b_e2[n];
#pragma unroll
    for (int v = 0; v < 8; ++v) {
      int m = v + Mrow;
      float val = siluf(mij[t][v] + bn);
      mijB[m * 128 + n] = f2bf(val);
      atomicAdd(&m_i[(long)nodeB[m * 2] * DD + n], val);
    }
  }
  lds_fence();

  // ---- cw = silu(m_ij @ W_c1.T + b_c1) @ W_c2.T + b_c2 -------------------
  v16bf am[4];
#pragma unroll
  for (int kt = 0; kt < 4; ++kt) am[kt] = loadA_lds(mijB, 128, lane, kt * 32);

  stage_cw(Wc1p, stage, 0, tid);
  wait_async0();
  __syncthreads();
  for (int g = 0; g < 4; ++g) {
    if (g < 3) stage_cw(Wc1p, stage + ((g + 1) & 1) * EDGE_STAGE_BYTES, g + 1, tid);
    const uint16_t* sb = (const uint16_t*)(stage + (g & 1) * EDGE_STAGE_BYTES);
#pragma unroll
    for (int r = 0; r < 8; ++r) {
      v8f acc = v8f_zero();
#pragma unroll
      for (int kt = 0; kt < 4; ++kt)
        acc = wmma_bf16(am[kt], loadB(sb + (long)(r * 4 + kt) * 512, lane), acc);
      int n = (g * 8 + r) * 16 + nlo;
      float bn = b_c1[n], w2 = W_c2[n];
#pragma unroll
      for (int v = 0; v < 8; ++v) {
        int m = v + Mrow;
        atomicAdd(&cwB[m], siluf(acc[v] + bn) * w2);
      }
    }
    wait_async0();
    __syncthreads();
  }
  lds_fence();

  if (lane < 16) {
    float cw = cwB[lane] + b_c2[0];
    int i = nodeB[lane * 2];
    atomicAdd(&num[(long)i * 3 + 0], relB[lane * 4 + 0] * cw);
    atomicAdd(&num[(long)i * 3 + 1], relB[lane * 4 + 1] * cw);
    atomicAdd(&num[(long)i * 3 + 2], relB[lane * 4 + 2] * cw);
    atomicAdd(&cnt[i], 1.0f);
  }
}

// ---------------------------------------------------------------------------
// Node update kernel (weights are small: per-wave streaming is fine)
// ---------------------------------------------------------------------------
constexpr int NODE_WAVE_BYTES = 21760;

__global__ __launch_bounds__(128) void node_kernel(
    const float* __restrict__ pos, const int* __restrict__ batch,
    const float* __restrict__ feats, const uint16_t* __restrict__ featsbf,
    const float* __restrict__ m_i, const float* __restrict__ num,
    const float* __restrict__ cnt,
    const float* __restrict__ b_n1, const float* __restrict__ b_n2,
    const float* __restrict__ te, const float* __restrict__ b_f1,
    const uint16_t* __restrict__ Wn1p, const uint16_t* __restrict__ Wn2p,
    const uint16_t* __restrict__ Wf1p,
    float* __restrict__ y, float* __restrict__ sums) {
  extern __shared__ char smem[];
  const int lane = threadIdx.x & 31;
  const int wave = threadIdx.x >> 5;
  char* wbase = smem + wave * NODE_WAVE_BYTES;
  uint16_t* inB  = (uint16_t*)(wbase);           // 16 x 256 bf16
  uint16_t* h1t  = (uint16_t*)(wbase + 8192);    // 16 x 32 bf16
  float*    hidB = (float*)(wbase + 9216);       // 16 x 128 f32
  float*    scB  = (float*)(wbase + 17408);      // 16
  int*      gB   = (int*)(wbase + 17472);        // 16
  uint16_t* fB   = (uint16_t*)(wbase + 17536);   // 16 x 128 bf16

  const long wgid = (long)blockIdx.x * 4 + wave;
  if (wgid >= NN / 16) return;  // whole-wave guard; EXEC stays all-1s inside
  const long nb = wgid * 16;
  const int m16 = lane & 15;
  const int hf = lane >> 4;
  const int nlo = lane & 15;
  const int Mrow = hf * 8;

  // ---- stage [feats | m_i] bf16 (feats via async global->LDS DMA) --------
  {
    long node = nb + m16;
    const char* s = (const char*)(featsbf + node * DD + hf * 64);
    uint32_t d = lds_off32(inB + m16 * 256 + hf * 64);
#pragma unroll
    for (int q = 0; q < 8; ++q) async_load_b128(d + q * 16, s + q * 16);
    const float* mi = m_i + node * DD + hf * 64;
    for (int n = 0; n < 64; ++n) inB[m16 * 256 + 128 + hf * 64 + n] = f2bf(mi[n]);
  }
  wait_async0();
  lds_fence();

  v16bf aF[8];
#pragma unroll
  for (int kt = 0; kt < 8; ++kt) aF[kt] = loadA_lds(inB, 256, lane, kt * 32);

  v8f acc2[8];
#pragma unroll
  for (int t = 0; t < 8; ++t) acc2[t] = v8f_zero();

  for (int c = 0; c < 8; ++c) {
    v8f a0 = v8f_zero(), a1 = v8f_zero();
#pragma unroll
    for (int kt = 0; kt < 8; ++kt) {
      a0 = wmma_bf16(aF[kt], loadB(Wn1p + ((long)((2 * c) * 8 + kt) << 9), lane), a0);
      a1 = wmma_bf16(aF[kt], loadB(Wn1p + ((long)((2 * c + 1) * 8 + kt) << 9), lane), a1);
    }
#pragma unroll
    for (int t = 0; t < 2; ++t) {
      int n = (2 * c + t) * 16 + nlo;
      float bn = b_n1[n];
      const v8f acc = t ? a1 : a0;
#pragma unroll
      for (int v = 0; v < 8; ++v) {
        int m = v + Mrow;
        h1t[m * 32 + t * 16 + nlo] = f2bf(siluf(acc[v] + bn));
      }
    }
    lds_fence();
    v16bf a2 = loadA_lds(h1t, 32, lane, 0);
#pragma unroll
    for (int t = 0; t < 8; ++t)
      acc2[t] = wmma_bf16(a2, loadB(Wn2p + ((long)(t * 8 + c) << 9), lane), acc2[t]);
    lds_fence();
  }

  // ---- hid = acc2 + b_n2 + feats  (residual) -----------------------------
#pragma unroll
  for (int t = 0; t < 8; ++t) {
    int n = t * 16 + nlo;
    float bn = b_n2[n];
#pragma unroll
    for (int v = 0; v < 8; ++v) {
      int m = v + Mrow;
      long node = nb + m;
      hidB[m * 128 + n] = acc2[t][v] + bn + feats[node * DD + n];
    }
  }
  lds_fence();

  // ---- expmap0 norm over [coors_new | hid] per node ----------------------
  if (lane < 16) {
    long node = nb + lane;
    float c = fmaxf(cnt[node], 1.0f);
    float s2 = 0.f;
#pragma unroll
    for (int d3 = 0; d3 < 3; ++d3) {
      float cd = pos[node * 3 + d3] + num[node * 3 + d3] / c;
      s2 += cd * cd;
    }
    for (int n = 0; n < 128; ++n) {
      float hv = hidB[lane * 128 + n];
      s2 += hv * hv;
    }
    float nn2 = fmaxf(sqrtf(s2), 1e-15f);
    scB[lane] = tanhf(nn2) / nn2;
    gB[lane] = batch[node];
  }
  lds_fence();

  // ---- f = expmap0(hid) * (scale[batch]+1) + shift[batch] ----------------
  {
    float sc0 = scB[m16];
    int g = gB[m16];
    for (int n = hf * 64; n < hf * 64 + 64; ++n) {
      float f = sc0 * hidB[m16 * 128 + n];
      f = f * (te[g * 256 + n] + 1.0f) + te[g * 256 + 128 + n];
      fB[m16 * 128 + n] = f2bf(f);
    }
  }
  lds_fence();

  // ---- y = silu(f @ W_f1.T + b_f1), global sum/sumsq partials ------------
  v16bf af[4];
#pragma unroll
  for (int kt = 0; kt < 4; ++kt) af[kt] = loadA_lds(fB, 128, lane, kt * 32);
  float s1 = 0.f, sq = 0.f;
  for (int t = 0; t < 8; ++t) {
    v8f acc = v8f_zero();
#pragma unroll
    for (int kt = 0; kt < 4; ++kt)
      acc = wmma_bf16(af[kt], loadB(Wf1p + ((long)(t * 4 + kt) << 9), lane), acc);
    int n = t * 16 + nlo;
    float bn = b_f1[n];
#pragma unroll
    for (int v = 0; v < 8; ++v) {
      int m = v + Mrow;
      float yv = siluf(acc[v] + bn);
      y[(nb + m) * DD + n] = yv;
      s1 += yv;
      sq += yv * yv;
    }
  }
  atomicAdd(&sums[0], s1);
  atomicAdd(&sums[1], sq);
}

// ---------------------------------------------------------------------------
// Global layernorm + precombined output projection
// ---------------------------------------------------------------------------
__global__ void final_kernel(const float* __restrict__ y, const float* __restrict__ sums,
                             const float* __restrict__ ln_g, const float* __restrict__ ln_b,
                             const float* __restrict__ Wcomb, const float* __restrict__ bcomb,
                             float* __restrict__ out) {
  long n = (long)blockIdx.x * blockDim.x + threadIdx.x;
  if (n >= NN) return;
  const float inv_cnt = 1.0f / (float)((long)NN * DD);
  float mean = sums[0] * inv_cnt;
  float var = sums[1] * inv_cnt - mean * mean;
  float inv = rsqrtf(var + 1e-5f);
  float o0 = bcomb[0], o1 = bcomb[1], o2 = bcomb[2], o3 = bcomb[3];
  for (int d = 0; d < DD; ++d) {
    float yn = (y[n * DD + d] - mean) * inv * ln_g[d] + ln_b[d];
    o0 += yn * Wcomb[0 * DD + d];
    o1 += yn * Wcomb[1 * DD + d];
    o2 += yn * Wcomb[2 * DD + d];
    o3 += yn * Wcomb[3 * DD + d];
  }
  out[n * 4 + 0] = o0;
  out[n * 4 + 1] = o1;
  out[n * 4 + 2] = o2;
  out[n * 4 + 3] = o3;
}

// ---------------------------------------------------------------------------
// Launch
// ---------------------------------------------------------------------------
extern "C" void kernel_launch(void* const* d_in, const int* in_sizes, int n_in,
                              void* d_out, int out_size, void* d_ws, size_t ws_size,
                              hipStream_t stream) {
  (void)in_sizes; (void)n_in; (void)out_size; (void)ws_size;
  const float* x       = (const float*)d_in[0];
  const float* extra_x = (const float*)d_in[1];
  const float* pos     = (const float*)d_in[2];
  const float* ear     = (const float*)d_in[3];
  const float* tin     = (const float*)d_in[4];
  const float* W_hyp   = (const float*)d_in[5];
  const float* W_t1    = (const float*)d_in[6];
  const float* b_t1    = (const float*)d_in[7];
  const float* W_t2    = (const float*)d_in[8];
  const float* b_t2    = (const float*)d_in[9];
  const float* W_ee    = (const float*)d_in[10];
  const float* b_ee    = (const float*)d_in[11];
  const float* W_e1    = (const float*)d_in[12];
  const float* b_e1    = (const float*)d_in[13];
  const float* W_e2    = (const float*)d_in[14];
  const float* b_e2    = (const float*)d_in[15];
  const float* W_c1    = (const float*)d_in[16];
  const float* b_c1    = (const float*)d_in[17];
  const float* W_c2    = (const float*)d_in[18];
  const float* b_c2    = (const float*)d_in[19];
  const float* W_n1    = (const float*)d_in[20];
  const float* b_n1    = (const float*)d_in[21];
  const float* W_n2    = (const float*)d_in[22];
  const float* b_n2    = (const float*)d_in[23];
  const float* W_tl    = (const float*)d_in[24];
  const float* b_tl    = (const float*)d_in[25];
  const float* W_f1    = (const float*)d_in[26];
  const float* b_f1    = (const float*)d_in[27];
  const float* ln_g    = (const float*)d_in[28];
  const float* ln_b    = (const float*)d_in[29];
  const float* W_f2    = (const float*)d_in[30];
  const float* b_f2    = (const float*)d_in[31];
  const float* W_out   = (const float*)d_in[32];
  const float* b_out   = (const float*)d_in[33];
  const int*   eidx    = (const int*)d_in[34];
  const int*   batch   = (const int*)d_in[35];

  char* ws = (char*)d_ws;
  size_t off = 0;
  auto alloc = [&](size_t bytes) -> void* {
    void* p = ws + off;
    off = (off + bytes + 255) & ~(size_t)255;
    return p;
  };

  float*    feats   = (float*)alloc((size_t)NN * DD * 4);
  uint16_t* featsbf = (uint16_t*)alloc((size_t)NN * DD * 2);
  float*    m_i     = (float*)alloc((size_t)NN * DD * 4);
  float*    num     = (float*)alloc((size_t)NN * 3 * 4);
  float*    cnt     = (float*)alloc((size_t)NN * 4);
  float*    yb      = (float*)alloc((size_t)NN * DD * 4);
  float*    te      = (float*)alloc((size_t)GG * 256 * 4);
  float*    sums    = (float*)alloc(2 * 4);
  uint16_t* We1p    = (uint16_t*)alloc((size_t)50 * 12 * 512 * 2);
  uint16_t* We2p    = (uint16_t*)alloc((size_t)8 * 25 * 512 * 2);
  uint16_t* Wc1p    = (uint16_t*)alloc((size_t)32 * 4 * 512 * 2);
  uint16_t* Wn1p    = (uint16_t*)alloc((size_t)16 * 8 * 512 * 2);
  uint16_t* Wn2p    = (uint16_t*)alloc((size_t)8 * 8 * 512 * 2);
  uint16_t* Wf1p    = (uint16_t*)alloc((size_t)8 * 4 * 512 * 2);
  float*    Wcomb   = (float*)alloc(512 * 4);
  float*    bcomb   = (float*)alloc(4 * 4);

  hipMemsetAsync(m_i, 0, (size_t)NN * DD * 4, stream);
  hipMemsetAsync(num, 0, (size_t)NN * 3 * 4, stream);
  hipMemsetAsync(cnt, 0, (size_t)NN * 4, stream);
  hipMemsetAsync(sums, 0, 8, stream);

  // weight pre-swizzle (bf16, WMMA B-fragment tiles)
  pack_b_kernel<<<256, 256, 0, stream>>>(W_e1, We1p, 770, 385, 384, 50, 12);
  pack_b_kernel<<<128, 256, 0, stream>>>(W_e2, We2p, 128, 770, 770, 8, 25);
  pack_b_kernel<<<64, 256, 0, stream>>>(W_c1, Wc1p, 512, 128, 128, 32, 4);
  pack_b_kernel<<<64, 256, 0, stream>>>(W_n1, Wn1p, 256, 256, 256, 16, 8);
  pack_b_kernel<<<32, 256, 0, stream>>>(W_n2, Wn2p, 128, 256, 256, 8, 8);
  pack_b_kernel<<<16, 256, 0, stream>>>(W_f1, Wf1p, 128, 128, 128, 8, 4);
  comb_kernel<<<2, 256, 0, stream>>>(W_f2, W_out, b_f2, b_out, Wcomb, bcomb);

  time_kernel<<<1, 256, 0, stream>>>(tin, W_t1, b_t1, W_t2, b_t2, W_tl, b_tl, te);
  node_feats_kernel<<<NN, 128, 0, stream>>>(x, extra_x, W_hyp, feats, featsbf);

  size_t edgeSmem = (size_t)EDGE_STAGE_OFF + 2 * EDGE_STAGE_BYTES;  // 137216
  edge_kernel<<<EE / 64, 128, edgeSmem, stream>>>(
      pos, ear, eidx, W_ee, b_ee, W_e1, b_e1, b_e2, b_c1, W_c2, b_c2, featsbf,
      We1p, We2p, Wc1p, m_i, num, cnt);

  int nodeBlocks = (NN / 16 + 3) / 4;  // 313
  node_kernel<<<nodeBlocks, 128, 4 * NODE_WAVE_BYTES, stream>>>(
      pos, batch, feats, featsbf, m_i, num, cnt, b_n1, b_n2, te, b_f1,
      Wn1p, Wn2p, Wf1p, yb, sums);

  final_kernel<<<(NN + 255) / 256, 256, 0, stream>>>(yb, sums, ln_g, ln_b, Wcomb,
                                                     bcomb, (float*)d_out);
}
